// AttentionBlock_87892210745427
// MI455X (gfx1250) — matrix operand order repaired and checked by reading, not compiled, not run on hardware
//
#include <hip/hip_runtime.h>
#include <hip/hip_bf16.h>

// ---------------------------------------------------------------------------
// AttentionBlock for MI455X (gfx1250, wave32, WMMA bf16 16x16x32)
//   1) cvt weights fp32->bf16
//   2) groupnorm(x),(y) -> transposed bf16 xnT/ynT [b][t][c]
//   3) GEMM q = xnT*Wq^T -> qT; GEMM kv = ynT*Wkv^T -> kT + vT(transposed)
//   4) flash attention (no 512MB score tensor) -> aT
//   5) GEMM proj + bias + residual -> out (fp32)
// GEMMs are double-buffered so WMMAs never wait on same-iteration loads.
// ---------------------------------------------------------------------------

constexpr int kB  = 16;
constexpr int kC  = 512;
constexpr int kT  = 1024;
constexpr int kG  = 32;
constexpr int kCG = 16;   // channels per group
constexpr int kNH = 8;
constexpr int kCH = 64;   // head dim
constexpr int kBM = 128;  // t-rows per attention workgroup (8 waves x 16)
constexpr int kBN = 64;   // s-block size

typedef __attribute__((ext_vector_type(16))) __bf16 v16bf;
typedef __attribute__((ext_vector_type(8)))  float  v8f;

struct alignas(16) B128 { unsigned int x[4]; };

union Frag16 { v16bf v; B128 q[2]; };
union FragE  { v16bf v; __bf16 e[16]; };

__device__ __forceinline__ v8f wmma_bf16(v16bf a, v16bf b, v8f c) {
  return __builtin_amdgcn_wmma_f32_16x16x32_bf16(
      false, a, false, b, (short)0, c, false, false);
}

// A operand: row M=lane&15; elems[0..7]=K(klo..), elems[8..15]=K(klo+16..)
__device__ __forceinline__ v16bf load_a(const __bf16* base, int rowstride, int k0) {
  const int lane = threadIdx.x & 31;
  const __bf16* p = base + (size_t)(lane & 15) * rowstride + k0 + ((lane < 16) ? 0 : 8);
  Frag16 f;
  f.q[0] = *(const B128*)(p);
  f.q[1] = *(const B128*)(p + 16);
  return f.v;
}

// B operand: column N=lane&15; elems[0..7]=K(klo..), elems[8..15]=K(klo+16..)
__device__ __forceinline__ v16bf load_b(const __bf16* base, int rowstride, int k0) {
  const int lane = threadIdx.x & 31;
  const __bf16* p = base + (size_t)(lane & 15) * rowstride + k0 + ((lane < 16) ? 0 : 8);
  Frag16 f;
  f.q[0] = *(const B128*)(p);
  f.q[1] = *(const B128*)(p + 16);
  return f.v;
}

// A operand sourced from f32 LDS (softmax P tile), converted to bf16
__device__ __forceinline__ v16bf load_a_f32(const float* base, int rowstride, int k0) {
  const int lane = threadIdx.x & 31;
  const float* p = base + (lane & 15) * rowstride + k0 + ((lane < 16) ? 0 : 8);
  FragE f;
#pragma unroll
  for (int i = 0; i < 8; ++i) {
    f.e[i]     = (__bf16)p[i];
    f.e[i + 8] = (__bf16)p[16 + i];
  }
  return f.v;
}

__device__ __forceinline__ float redmax16(float v) {
#pragma unroll
  for (int m = 1; m < 16; m <<= 1) v = fmaxf(v, __shfl_xor(v, m, 32));
  return v;
}
__device__ __forceinline__ float redsum16(float v) {
#pragma unroll
  for (int m = 1; m < 16; m <<= 1) v += __shfl_xor(v, m, 32);
  return v;
}

// ============================ 1) fp32 -> bf16 ==============================
__global__ void cvt_bf16_kernel(const float* __restrict__ s, __bf16* __restrict__ d, int n) {
  int i = blockIdx.x * blockDim.x + threadIdx.x;
  if (i < n) d[i] = (__bf16)s[i];
}

// ============================ 2) GroupNorm =================================
__global__ void gn_kernel(const float* __restrict__ x, const float* __restrict__ w,
                          const float* __restrict__ bb, __bf16* __restrict__ outT) {
  const int bg = blockIdx.x;
  const int b = bg >> 5;
  const int g = bg & (kG - 1);
  const float* xp = x + ((size_t)b * kC + (size_t)g * kCG) * kT;  // contiguous 16384 floats

  float s = 0.f, s2 = 0.f;
  for (int i = threadIdx.x; i < kCG * kT; i += 256) {
    float v = xp[i];
    s += v; s2 += v * v;
  }
  __shared__ float ls[256], ls2[256];
  ls[threadIdx.x] = s; ls2[threadIdx.x] = s2;
  __syncthreads();
  for (int st = 128; st > 0; st >>= 1) {
    if (threadIdx.x < st) { ls[threadIdx.x] += ls[threadIdx.x + st];
                            ls2[threadIdx.x] += ls2[threadIdx.x + st]; }
    __syncthreads();
  }
  const float inv_n = 1.f / (float)(kCG * kT);
  const float mu   = ls[0] * inv_n;
  const float var  = ls2[0] * inv_n - mu * mu;
  const float rstd = rsqrtf(var + 1e-5f);

  const int cc = threadIdx.x & 15;
  const int c  = g * kCG + cc;
  const float wc = w[c], bc = bb[c];
  for (int t = (threadIdx.x >> 4); t < kT; t += 16) {
    float v = xp[(size_t)cc * kT + t];
    outT[((size_t)b * kT + t) * kC + c] = (__bf16)((v - mu) * rstd * wc + bc);
  }
}

// ============================ 3) q / kv GEMM ===============================
// D[t][o] = sum_c Xt[t][c]*W[o][c] + bias[o];  o>=Osplit stored transposed (V)
// Double-buffered: fragments for k+32 are in flight while k is consumed.
__global__ void gemm_qkv_kernel(const __bf16* __restrict__ Xt, const __bf16* __restrict__ W,
                                const float* __restrict__ bias, __bf16* __restrict__ outT,
                                __bf16* __restrict__ outVT, int Osplit) {
  const int lane = threadIdx.x & 31;
  const int wave = threadIdx.x >> 5;
  const int b  = blockIdx.z;
  const int t0 = blockIdx.x * 128 + wave * 16;
  const int o0 = blockIdx.y * 64;

  const __bf16* arow = Xt + ((size_t)b * kT + t0) * kC;
  v8f acc[4] = {};

  v16bf a_cur = load_a(arow, kC, 0);
  v16bf b_cur[4];
#pragma unroll
  for (int j = 0; j < 4; ++j)
    b_cur[j] = load_b(W + (size_t)(o0 + j * 16) * kC, kC, 0);

#pragma unroll 4
  for (int k0 = 0; k0 < kC; k0 += 32) {
    const int kp = (k0 + 32 < kC) ? (k0 + 32) : 0;   // last iter: harmless reload
    v16bf a_nxt = load_a(arow, kC, kp);
    v16bf b_nxt[4];
#pragma unroll
    for (int j = 0; j < 4; ++j)
      b_nxt[j] = load_b(W + (size_t)(o0 + j * 16) * kC, kC, kp);
#pragma unroll
    for (int j = 0; j < 4; ++j)
      acc[j] = wmma_bf16(a_cur, b_cur[j], acc[j]);
    a_cur = a_nxt;
#pragma unroll
    for (int j = 0; j < 4; ++j) b_cur[j] = b_nxt[j];
  }

  const int mbase = t0 + ((lane < 16) ? 0 : 8);
#pragma unroll
  for (int j = 0; j < 4; ++j) {
    const int o = o0 + j * 16 + (lane & 15);
    const float bv = bias[o];
#pragma unroll
    for (int r = 0; r < 8; ++r) {
      const int t = mbase + r;
      const float v = acc[j][r] + bv;
      if (o < Osplit) {
        outT[((size_t)b * kT + t) * Osplit + o] = (__bf16)v;
      } else {
        const int oo = o - Osplit;
        const int h = oo >> 6, d = oo & 63;
        outVT[(((size_t)b * kNH + h) * kCH + d) * kT + t] = (__bf16)v;
      }
    }
  }
}

// ============================ 4) flash attention ===========================
__global__ void attn_kernel(const __bf16* __restrict__ qT, const __bf16* __restrict__ kTp,
                            const __bf16* __restrict__ vTp, __bf16* __restrict__ aT) {
  __shared__ alignas(16) __bf16 sK[kBN * kCH];      // [s][d]
  __shared__ alignas(16) __bf16 sV[kCH * kBN];      // [d][s]
  __shared__ alignas(16) float  sP[8][16 * kBN];    // per-wave [m][s]

  const int lane = threadIdx.x & 31;
  const int wave = threadIdx.x >> 5;
  const int b = blockIdx.z, h = blockIdx.y;
  const int t0 = blockIdx.x * kBM + wave * 16;
  const size_t bh = (size_t)b * kNH + h;

  const __bf16* qbase = qT + ((size_t)b * kT + t0) * kC + h * kCH;
  v16bf qa0 = load_a(qbase, kC, 0);
  v16bf qa1 = load_a(qbase, kC, 32);

  float mrow[8], lrow[8];
  v8f oacc[4] = {};
#pragma unroll
  for (int r = 0; r < 8; ++r) { mrow[r] = -1e30f; lrow[r] = 0.f; }

  for (int s0 = 0; s0 < kT; s0 += kBN) {
    // stage K [s][d] and V [d][s] tiles (coalesced 16B chunks)
    const __bf16* kg = kTp + ((size_t)b * kT + s0) * kC + h * kCH;
    for (int i = threadIdx.x; i < (kBN * kCH) / 8; i += 256) {
      int row = i >> 3, qd = i & 7;
      *(B128*)&sK[row * kCH + qd * 8] = *(const B128*)(kg + (size_t)row * kC + qd * 8);
    }
    const __bf16* vg = vTp + (bh * kCH) * kT + s0;
    for (int i = threadIdx.x; i < (kCH * kBN) / 8; i += 256) {
      int row = i >> 3, qd = i & 7;
      *(B128*)&sV[row * kBN + qd * 8] = *(const B128*)(vg + (size_t)row * kT + qd * 8);
    }
    __syncthreads();

    // ---- S = (Q K^T)/8 : burst all 8 B-frag DS loads, then 8 WMMAs
    v16bf kb[8];
#pragma unroll
    for (int j = 0; j < 4; ++j) {
      kb[j * 2 + 0] = load_b(&sK[(j * 16) * kCH], kCH, 0);
      kb[j * 2 + 1] = load_b(&sK[(j * 16) * kCH], kCH, 32);
    }
    v8f sacc[4] = {};
#pragma unroll
    for (int j = 0; j < 4; ++j) {
      sacc[j] = wmma_bf16(qa0, kb[j * 2 + 0], sacc[j]);
      sacc[j] = wmma_bf16(qa1, kb[j * 2 + 1], sacc[j]);
    }

    float* Pw = &sP[wave][0];
    // ---- online softmax per row (rows live across 16-lane halves)
#pragma unroll
    for (int r = 0; r < 8; ++r) {
      float tm = -1e30f;
#pragma unroll
      for (int j = 0; j < 4; ++j) { sacc[j][r] *= 0.125f; tm = fmaxf(tm, sacc[j][r]); }
      tm = redmax16(tm);
      const float mnew  = fmaxf(mrow[r], tm);
      const float alpha = __expf(mrow[r] - mnew);
      mrow[r] = mnew;
      float psum = 0.f;
      const int m = r + ((lane < 16) ? 0 : 8);
#pragma unroll
      for (int j = 0; j < 4; ++j) {
        const float p = __expf(sacc[j][r] - mnew);
        psum += p;
        Pw[m * kBN + j * 16 + (lane & 15)] = p;
      }
      psum = redsum16(psum);
      lrow[r] = lrow[r] * alpha + psum;
#pragma unroll
      for (int j = 0; j < 4; ++j) oacc[j][r] *= alpha;
    }

    // ---- O += P * V : burst all loads, then 8 WMMAs
    v16bf pa0 = load_a_f32(Pw, kBN, 0);
    v16bf pa1 = load_a_f32(Pw, kBN, 32);
    v16bf vb[8];
#pragma unroll
    for (int j = 0; j < 4; ++j) {
      vb[j * 2 + 0] = load_b(&sV[(j * 16) * kBN], kBN, 0);
      vb[j * 2 + 1] = load_b(&sV[(j * 16) * kBN], kBN, 32);
    }
#pragma unroll
    for (int j = 0; j < 4; ++j) {
      oacc[j] = wmma_bf16(pa0, vb[j * 2 + 0], oacc[j]);
      oacc[j] = wmma_bf16(pa1, vb[j * 2 + 1], oacc[j]);
    }
    __syncthreads();
  }

#pragma unroll
  for (int j = 0; j < 4; ++j) {
    const int d = j * 16 + (lane & 15);
#pragma unroll
    for (int r = 0; r < 8; ++r) {
      const int t = t0 + r + ((lane < 16) ? 0 : 8);
      aT[((size_t)b * kT + t) * kC + h * kCH + d] = (__bf16)(oacc[j][r] / lrow[r]);
    }
  }
}

// ============================ 5) projection + residual =====================
__global__ void gemm_proj_kernel(const __bf16* __restrict__ Wpb, const __bf16* __restrict__ aTb,
                                 const float* __restrict__ bp, const float* __restrict__ x,
                                 float* __restrict__ out) {
  const int lane = threadIdx.x & 31;
  const int wave = threadIdx.x >> 5;
  const int b  = blockIdx.z;
  const int t0 = blockIdx.x * 64;
  const int o0 = blockIdx.y * 128 + wave * 16;

  const __bf16* arow  = Wpb + (size_t)o0 * kC;
  const __bf16* bbase = aTb + ((size_t)b * kT + t0) * kC;
  v8f acc[4] = {};

  v16bf a_cur = load_a(arow, kC, 0);
  v16bf b_cur[4];
#pragma unroll
  for (int j = 0; j < 4; ++j)
    b_cur[j] = load_b(bbase + (size_t)(j * 16) * kC, kC, 0);

#pragma unroll 4
  for (int k0 = 0; k0 < kC; k0 += 32) {
    const int kp = (k0 + 32 < kC) ? (k0 + 32) : 0;
    v16bf a_nxt = load_a(arow, kC, kp);
    v16bf b_nxt[4];
#pragma unroll
    for (int j = 0; j < 4; ++j)
      b_nxt[j] = load_b(bbase + (size_t)(j * 16) * kC, kC, kp);
#pragma unroll
    for (int j = 0; j < 4; ++j)
      acc[j] = wmma_bf16(a_cur, b_cur[j], acc[j]);
    a_cur = a_nxt;
#pragma unroll
    for (int j = 0; j < 4; ++j) b_cur[j] = b_nxt[j];
  }

#pragma unroll
  for (int r = 0; r < 8; ++r) {
    const int o = o0 + r + ((lane < 16) ? 0 : 8);
    const float bv = bp[o];
#pragma unroll
    for (int j = 0; j < 4; ++j) {
      const int t = t0 + j * 16 + (lane & 15);
      const size_t idx = ((size_t)b * kC + o) * kT + t;
      out[idx] = acc[j][r] + bv + x[idx];
    }
  }
}

// ===========================================================================
extern "C" void kernel_launch(void* const* d_in, const int* in_sizes, int n_in,
                              void* d_out, int out_size, void* d_ws, size_t ws_size,
                              hipStream_t stream) {
  (void)in_sizes; (void)n_in; (void)out_size; (void)ws_size;
  const float* x    = (const float*)d_in[0];
  const float* y    = (const float*)d_in[1];
  const float* gn_w = (const float*)d_in[2];
  const float* gn_b = (const float*)d_in[3];
  const float* Wq   = (const float*)d_in[4];
  const float* bq   = (const float*)d_in[5];
  const float* Wkv  = (const float*)d_in[6];
  const float* bkv  = (const float*)d_in[7];
  const float* Wp   = (const float*)d_in[8];
  const float* bp   = (const float*)d_in[9];
  float* out = (float*)d_out;

  size_t off = 0;
  char* wsb = (char*)d_ws;
  auto take = [&](size_t bytes) -> char* {
    char* p = wsb + off;
    off += (bytes + 255) & ~(size_t)255;
    return p;
  };
  const size_t btc = (size_t)kB * kT * kC * sizeof(__bf16);
  __bf16* xnT  = (__bf16*)take(btc);
  __bf16* ynT  = (__bf16*)take(btc);
  __bf16* WqB  = (__bf16*)take((size_t)kC * kC * sizeof(__bf16));
  __bf16* WkvB = (__bf16*)take((size_t)2 * kC * kC * sizeof(__bf16));
  __bf16* WpB  = (__bf16*)take((size_t)kC * kC * sizeof(__bf16));
  __bf16* qTb  = (__bf16*)take(btc);
  __bf16* kTb  = (__bf16*)take(btc);
  __bf16* vTb  = (__bf16*)take(btc);
  __bf16* aTb  = (__bf16*)take(btc);

  {
    int n1 = kC * kC;
    cvt_bf16_kernel<<<(n1 + 255) / 256, 256, 0, stream>>>(Wq, WqB, n1);
    int n2 = 2 * kC * kC;
    cvt_bf16_kernel<<<(n2 + 255) / 256, 256, 0, stream>>>(Wkv, WkvB, n2);
    cvt_bf16_kernel<<<(n1 + 255) / 256, 256, 0, stream>>>(Wp, WpB, n1);
  }
  gn_kernel<<<kB * kG, 256, 0, stream>>>(x, gn_w, gn_b, xnT);
  gn_kernel<<<kB * kG, 256, 0, stream>>>(y, gn_w, gn_b, ynT);

  gemm_qkv_kernel<<<dim3(kT / 128, kC / 64, kB), 256, 0, stream>>>(
      xnT, WqB, bq, qTb, nullptr, kC);
  gemm_qkv_kernel<<<dim3(kT / 128, (2 * kC) / 64, kB), 256, 0, stream>>>(
      ynT, WkvB, bkv, kTb, vTb, kC);

  attn_kernel<<<dim3(kT / kBM, kNH, kB), 256, 0, stream>>>(qTb, kTb, vTb, aTb);

  gemm_proj_kernel<<<dim3(kT / 64, kC / 128, kB), 256, 0, stream>>>(WpB, aTb, bp, x, out);
}
